// Classifier_18605798326559
// MI455X (gfx1250) — compile-verified, matrix operand order checked by reading, not run on hardware
//
#include <hip/hip_runtime.h>
#include <hip/hip_bf16.h>
#include <math.h>

typedef __attribute__((ext_vector_type(8)))  __bf16 v8bf;
typedef __attribute__((ext_vector_type(16))) __bf16 v16bf;
typedef __attribute__((ext_vector_type(8)))  float  v8f;

#define B_ROWS 8192
#define H_DIM  2048
#define C_CLS  1000
#define NPAD   1024
#define GAMMA_ 100.0f
#define EPS_   1e-8f

// LDS row stride in ushorts: 40 (80 bytes) -> 16B-aligned chunks, conflict-free b128 loads
#define LSTR 40

#if defined(__has_builtin)
#if __has_builtin(__builtin_amdgcn_global_load_async_to_lds_b128)
#define HAVE_ASYNC_LDS 1
#endif
#if __has_builtin(__builtin_amdgcn_s_wait_asynccnt)
#define HAVE_WAIT_ASYNC_BUILTIN 1
#endif
#endif

#if defined(HAVE_ASYNC_LDS)
// Parameter types per hipcc diagnostic: ptr to 16B int vector, global (AS1) src / LDS (AS3) dst
typedef int async_v4i __attribute__((vector_size(16)));
typedef __attribute__((address_space(1))) async_v4i* g_async_ptr;
typedef __attribute__((address_space(3))) async_v4i* l_async_ptr;
#if defined(HAVE_WAIT_ASYNC_BUILTIN)
#define WAIT_ASYNCCNT0() __builtin_amdgcn_s_wait_asynccnt(0)
#else
#define WAIT_ASYNCCNT0() asm volatile("s_wait_asynccnt 0x0" ::: "memory")
#endif
#endif

union Frag16 { v16bf f; v8bf h[2]; };

static __device__ __forceinline__ void split_bf16(float x, unsigned short& hi, unsigned short& lo) {
    unsigned u  = __float_as_uint(x);
    unsigned hu = u & 0xFFFF0000u;
    hi = (unsigned short)(u >> 16);            // truncate-to-bf16
    float r = x - __uint_as_float(hu);         // exact residual
    lo = (unsigned short)(__float_as_uint(r) >> 16);
}

static __device__ __forceinline__ float block_sum256(float v, float* red) {
    __syncthreads();
    #pragma unroll
    for (int m = 16; m >= 1; m >>= 1) v += __shfl_xor(v, m, 32);
    const int w = threadIdx.x >> 5;
    if ((threadIdx.x & 31) == 0) red[w] = v;
    __syncthreads();
    if (threadIdx.x < 32) {
        float t = (threadIdx.x < 8) ? red[threadIdx.x] : 0.0f;
        #pragma unroll
        for (int m = 4; m >= 1; m >>= 1) t += __shfl_xor(t, m, 32);
        if (threadIdx.x == 0) red[0] = t;
    }
    __syncthreads();
    return red[0];
}

static __device__ __forceinline__ float block_max256(float v, float* red) {
    __syncthreads();
    #pragma unroll
    for (int m = 16; m >= 1; m >>= 1) v = fmaxf(v, __shfl_xor(v, m, 32));
    const int w = threadIdx.x >> 5;
    if ((threadIdx.x & 31) == 0) red[w] = v;
    __syncthreads();
    if (threadIdx.x < 32) {
        float t = (threadIdx.x < 8) ? red[threadIdx.x] : -INFINITY;
        #pragma unroll
        for (int m = 4; m >= 1; m >>= 1) t = fmaxf(t, __shfl_xor(t, m, 32));
        if (threadIdx.x == 0) red[0] = t;
    }
    __syncthreads();
    return red[0];
}

// --- Kernel 1: grp f32 -> bf16 hi/lo (padded to NPAD rows) + g_sq --------------------
__global__ __launch_bounds__(256) void prep_grp(const float* __restrict__ grp,
                                                unsigned short* __restrict__ Ghi,
                                                unsigned short* __restrict__ Glo,
                                                float* __restrict__ g_sq) {
    __shared__ float red[8];
    const int c   = blockIdx.x;
    const int tid = threadIdx.x;
    float s = 0.0f;
    if (c < C_CLS) {
        #pragma unroll
        for (int p = 0; p < 2; ++p) {
            const int idx = (p * 256 + tid) * 4;                 // 2048 floats / block
            const float4 v = *(const float4*)(grp + (size_t)c * H_DIM + idx);
            float f[4] = {v.x, v.y, v.z, v.w};
            unsigned short h[4], l[4];
            #pragma unroll
            for (int e = 0; e < 4; ++e) { split_bf16(f[e], h[e], l[e]); s += f[e] * f[e]; }
            uint2 hp, lp;
            hp.x = (unsigned)h[0] | ((unsigned)h[1] << 16);
            hp.y = (unsigned)h[2] | ((unsigned)h[3] << 16);
            lp.x = (unsigned)l[0] | ((unsigned)l[1] << 16);
            lp.y = (unsigned)l[2] | ((unsigned)l[3] << 16);
            *(uint2*)(&Ghi[(size_t)c * H_DIM + idx]) = hp;
            *(uint2*)(&Glo[(size_t)c * H_DIM + idx]) = lp;
        }
    } else {
        const uint2 z = make_uint2(0u, 0u);
        #pragma unroll
        for (int p = 0; p < 2; ++p) {
            const int idx = (p * 256 + tid) * 4;
            *(uint2*)(&Ghi[(size_t)c * H_DIM + idx]) = z;
            *(uint2*)(&Glo[(size_t)c * H_DIM + idx]) = z;
        }
    }
    const float tot = block_sum256(s, red);
    if (tid == 0) g_sq[c] = tot;
}

// --- Kernel 2: x_sq -------------------------------------------------------------------
__global__ __launch_bounds__(256) void prep_xsq(const float* __restrict__ X,
                                                float* __restrict__ x_sq) {
    __shared__ float red[8];
    const int b   = blockIdx.x;
    const int tid = threadIdx.x;
    float s = 0.0f;
    #pragma unroll
    for (int p = 0; p < 2; ++p) {
        const int idx = (p * 256 + tid) * 4;
        const float4 v = *(const float4*)(X + (size_t)b * H_DIM + idx);
        s += v.x * v.x + v.y * v.y + v.z * v.z + v.w * v.w;
    }
    const float tot = block_sum256(s, red);
    if (tid == 0) x_sq[b] = tot;
}

// --- Kernel 3: bf16x3 WMMA GEMM, dot -> d_out (cols < 1000) ---------------------------
__global__ __launch_bounds__(256) void gemm_kernel(const float* __restrict__ X,
                                                   const unsigned short* __restrict__ Ghi,
                                                   const unsigned short* __restrict__ Glo,
                                                   float* __restrict__ out) {
    __shared__ unsigned short sAh[128 * LSTR];
    __shared__ unsigned short sAl[128 * LSTR];
    __shared__ unsigned short sBh[128 * LSTR];
    __shared__ unsigned short sBl[128 * LSTR];

    const int tid = threadIdx.x;
    const int bm0 = blockIdx.x * 128;
    const int bn0 = blockIdx.y * 128;

    const int wave = tid >> 5;
    const int lane = tid & 31;
    const int lr   = lane & 15;
    const int lh   = lane >> 4;
    const int wm   = wave >> 2;   // 0..1 : 64 rows each
    const int wn   = wave & 3;    // 0..3 : 32 cols each

    v8f acc[4][2];
    #pragma unroll
    for (int i = 0; i < 4; ++i)
        #pragma unroll
        for (int j = 0; j < 2; ++j) acc[i][j] = (v8f)0.0f;

    const int arow = tid >> 3;    // 0..31
    const int akc  = tid & 7;     // float4 chunk within 32-wide K slab
    const int brow = tid >> 2;    // 0..63
    const int bkc  = tid & 3;     // 8-ushort chunk

    for (int kt = 0; kt < H_DIM / 32; ++kt) {
        const int k0 = kt * 32;
        __syncthreads();   // previous stage fully consumed before LDS overwrite

        // B: 128x32 bf16 hi/lo tiles -> LDS (2 passes of 64 rows).
        // Preferred path: gfx1250 async global->LDS DMA (ASYNCcnt-tracked, no VGPR round-trip).
        #pragma unroll
        for (int p = 0; p < 2; ++p) {
            const int r = p * 64 + brow;
            const size_t goff = (size_t)(bn0 + r) * H_DIM + k0 + bkc * 8;
            const int    loff = r * LSTR + bkc * 8;
#if defined(HAVE_ASYNC_LDS)
            __builtin_amdgcn_global_load_async_to_lds_b128(
                (g_async_ptr)(Ghi + goff), (l_async_ptr)(&sBh[loff]), 0, 0);
            __builtin_amdgcn_global_load_async_to_lds_b128(
                (g_async_ptr)(Glo + goff), (l_async_ptr)(&sBl[loff]), 0, 0);
#else
            const uint4 hv = *(const uint4*)(Ghi + goff);
            const uint4 lv = *(const uint4*)(Glo + goff);
            *(uint4*)(&sBh[loff]) = hv;
            *(uint4*)(&sBl[loff]) = lv;
#endif
        }

        // A: 128x32 f32 -> split bf16 hi/lo into LDS (4 passes of 32 rows)
        #pragma unroll
        for (int p = 0; p < 4; ++p) {
            const int r = p * 32 + arow;
            const float4 v = *(const float4*)(X + (size_t)(bm0 + r) * H_DIM + k0 + akc * 4);
            float f[4] = {v.x, v.y, v.z, v.w};
            unsigned short h[4], l[4];
            #pragma unroll
            for (int e = 0; e < 4; ++e) split_bf16(f[e], h[e], l[e]);
            uint2 hp, lp;
            hp.x = (unsigned)h[0] | ((unsigned)h[1] << 16);
            hp.y = (unsigned)h[2] | ((unsigned)h[3] << 16);
            lp.x = (unsigned)l[0] | ((unsigned)l[1] << 16);
            lp.y = (unsigned)l[2] | ((unsigned)l[3] << 16);
            *(uint2*)(&sAh[r * LSTR + akc * 4]) = hp;
            *(uint2*)(&sAl[r * LSTR + akc * 4]) = lp;
        }

#if defined(HAVE_ASYNC_LDS)
        WAIT_ASYNCCNT0();   // drain this wave's async-to-LDS before signaling barrier
#endif
        __syncthreads();

        // B fragments resident for this K-step
        Frag16 bh[2], bl[2];
        #pragma unroll
        for (int nt = 0; nt < 2; ++nt) {
            const int rB   = wn * 32 + nt * 16 + lr;
            const int base = rB * LSTR + lh * 8;   // ushort units
            bh[nt].h[0] = *(const v8bf*)(&sBh[base]);
            bh[nt].h[1] = *(const v8bf*)(&sBh[base + 16]);
            bl[nt].h[0] = *(const v8bf*)(&sBl[base]);
            bl[nt].h[1] = *(const v8bf*)(&sBl[base + 16]);
        }
        #pragma unroll
        for (int mt = 0; mt < 4; ++mt) {
            const int rA   = wm * 64 + mt * 16 + lr;
            const int base = rA * LSTR + lh * 8;
            Frag16 ah, al;
            ah.h[0] = *(const v8bf*)(&sAh[base]);
            ah.h[1] = *(const v8bf*)(&sAh[base + 16]);
            al.h[0] = *(const v8bf*)(&sAl[base]);
            al.h[1] = *(const v8bf*)(&sAl[base + 16]);
            #pragma unroll
            for (int nt = 0; nt < 2; ++nt) {
                acc[mt][nt] = __builtin_amdgcn_wmma_f32_16x16x32_bf16(
                    false, al.f, false, bh[nt].f, (short)0, acc[mt][nt], false, false);
                acc[mt][nt] = __builtin_amdgcn_wmma_f32_16x16x32_bf16(
                    false, ah.f, false, bl[nt].f, (short)0, acc[mt][nt], false, false);
                acc[mt][nt] = __builtin_amdgcn_wmma_f32_16x16x32_bf16(
                    false, ah.f, false, bh[nt].f, (short)0, acc[mt][nt], false, false);
            }
        }
    }

    // Writeout: C layout = lanes 0-15: VGPR e -> (M=e, N=lr); lanes 16-31: (M=e+8, N=lr)
    #pragma unroll
    for (int mt = 0; mt < 4; ++mt) {
        #pragma unroll
        for (int nt = 0; nt < 2; ++nt) {
            const int col = bn0 + wn * 32 + nt * 16 + lr;
            if (col < C_CLS) {
                #pragma unroll
                for (int e = 0; e < 8; ++e) {
                    const int row = bm0 + wm * 64 + mt * 16 + e + lh * 8;
                    out[(size_t)row * C_CLS + col] = acc[mt][nt][e];
                }
            }
        }
    }
}

// --- Kernel 4: fused cosine-softmax * euclid-softmin, in place on d_out ---------------
__global__ __launch_bounds__(256) void softmax_kernel(float* __restrict__ out,
                                                      const float* __restrict__ x_sq,
                                                      const float* __restrict__ g_sq) {
    __shared__ float red[8];
    const int b   = blockIdx.x;
    const int tid = threadIdx.x;
    const float xs = x_sq[b];
    const float xn = sqrtf(xs);

    float s1[4], s2[4];
    float m1 = -INFINITY, m2 = -INFINITY;
    #pragma unroll
    for (int i = 0; i < 4; ++i) {
        const int c = tid + i * 256;
        if (c < C_CLS) {
            const float dot = out[(size_t)b * C_CLS + c];
            const float gs  = g_sq[c];
            const float den = fmaxf(xn * sqrtf(gs), EPS_);
            const float v1  = GAMMA_ * dot / den;
            const float d2  = fmaxf(xs + gs - 2.0f * dot, 0.0f);
            const float v2  = -sqrtf(d2);
            s1[i] = v1; s2[i] = v2;
            m1 = fmaxf(m1, v1); m2 = fmaxf(m2, v2);
        } else { s1[i] = -INFINITY; s2[i] = -INFINITY; }
    }
    m1 = block_max256(m1, red);
    m2 = block_max256(m2, red);

    float e1[4], e2[4], t1 = 0.0f, t2 = 0.0f;
    #pragma unroll
    for (int i = 0; i < 4; ++i) {
        const int c = tid + i * 256;
        if (c < C_CLS) {
            e1[i] = expf(s1[i] - m1);
            e2[i] = expf(s2[i] - m2);
            t1 += e1[i]; t2 += e2[i];
        } else { e1[i] = 0.0f; e2[i] = 0.0f; }
    }
    t1 = block_sum256(t1, red);
    t2 = block_sum256(t2, red);
    const float inv = 1.0f / (t1 * t2);

    #pragma unroll
    for (int i = 0; i < 4; ++i) {
        const int c = tid + i * 256;
        if (c < C_CLS) out[(size_t)b * C_CLS + c] = e1[i] * e2[i] * inv;
    }
}

extern "C" void kernel_launch(void* const* d_in, const int* in_sizes, int n_in,
                              void* d_out, int out_size, void* d_ws, size_t ws_size,
                              hipStream_t stream) {
    const float* X   = (const float*)d_in[0];   // [8192, 2048] f32
    const float* grp = (const float*)d_in[1];   // [1000, 2048] f32
    float* out = (float*)d_out;                 // [8192, 1000] f32
    char* ws = (char*)d_ws;

    unsigned short* Ghi = (unsigned short*)ws;                                  // 4 MiB
    unsigned short* Glo = (unsigned short*)(ws + (size_t)4 * 1024 * 1024);      // 4 MiB
    float* g_sq = (float*)(ws + (size_t)8 * 1024 * 1024);                       // 4 KiB
    float* x_sq = (float*)(ws + (size_t)8 * 1024 * 1024 + 4096);                // 32 KiB

    prep_grp<<<NPAD, 256, 0, stream>>>(grp, Ghi, Glo, g_sq);
    prep_xsq<<<B_ROWS, 256, 0, stream>>>(X, x_sq);
    dim3 grid(B_ROWS / 128, NPAD / 128);   // 64 x 8
    gemm_kernel<<<grid, 256, 0, stream>>>(X, Ghi, Glo, out);
    softmax_kernel<<<B_ROWS, 256, 0, stream>>>(out, x_sq, g_sq);
}
